// GRU_5669356836107
// MI455X (gfx1250) — matrix operand order repaired and checked by reading, not compiled, not run on hardware
//
#include <hip/hip_runtime.h>
#include <hip/hip_bf16.h>

// ---------------------------------------------------------------------------
// GRU for MI455X (gfx1250): bf16 WMMA GEMMs with fp32 accumulate.
// Phase 0: convert x -> bf16, transpose kernel / recurrent_kernel -> bf16
// Phase 1: Wx = x @ kernel + bias. Register-blocked WMMA GEMM: each wave
//          owns a 32x64 output tile (2x4 sub-tiles, 8 fp32 accumulators),
//          so each 6 KB of L2 traffic feeds 8 v_wmma_f32_16x16x32_bf16
//          (~22 FLOP/byte vs 8 for the naive 1-tile version).
// Phase 2: 256 sequential step kernels, each: Rh = h @ R (WMMA, K split
//          over 4 waves, LDS reduce) then gate math + h update.
// ---------------------------------------------------------------------------

typedef __attribute__((ext_vector_type(16))) __bf16 v16bf;
typedef __attribute__((ext_vector_type(8)))  float  v8f;

static constexpr int T  = 256;
static constexpr int Bb = 64;
static constexpr int D  = 1024;
static constexpr int H  = 1024;
static constexpr int G3 = 3 * H;      // 3072

__device__ __forceinline__ __bf16 f2bf(float f) {
    unsigned u = __builtin_bit_cast(unsigned, f);
    u += 0x7FFFu + ((u >> 16) & 1u);               // round-to-nearest-even
    unsigned short h = (unsigned short)(u >> 16);
    return __builtin_bit_cast(__bf16, h);
}

__device__ __forceinline__ float sigmoidf(float x) {
    return 1.0f / (1.0f + __expf(-x));
}

// A tile (16xK slice, row-major bf16, ld = row stride in elements).
// Lane l (0..31): row = l&15 ; 16 contiguous K values at kb = k0 + (l>>4)*16.
__device__ __forceinline__ v16bf load_tile_A(const __bf16* __restrict__ base, int ld,
                                             int row0, int k0, int lane) {
    int r  = row0 + (lane & 15);
    int kb = k0 + ((lane >> 4) << 4);
    const __bf16* p = base + (size_t)r * ld + kb;
    union { v16bf v; uint4 q[2]; } u;
    u.q[0] = *reinterpret_cast<const uint4*>(p);
    u.q[1] = *reinterpret_cast<const uint4*>(p + 8);
    return u.v;
}

// B tile (Kx16 slice) from a pre-transposed matrix stored [N][K] bf16.
// Lane l: col = l&15 ; 16 contiguous K values at k0 + (l>>4)*16.
__device__ __forceinline__ v16bf load_tile_B(const __bf16* __restrict__ baseT, int ld,
                                             int col0, int k0, int lane) {
    int c  = col0 + (lane & 15);
    int kb = k0 + ((lane >> 4) << 4);
    const __bf16* p = baseT + (size_t)c * ld + kb;
    union { v16bf v; uint4 q[2]; } u;
    u.q[0] = *reinterpret_cast<const uint4*>(p);
    u.q[1] = *reinterpret_cast<const uint4*>(p + 8);
    return u.v;
}

// --------------------------- Phase 0 kernels -------------------------------

__global__ void __launch_bounds__(256) convert_x_kernel(const float* __restrict__ x,
                                                        __bf16* __restrict__ xb, int n) {
    int i = (blockIdx.x * 256 + threadIdx.x) * 4;
    if (i < n) {
        float4 v = *reinterpret_cast<const float4*>(x + i);
        union { __bf16 h[4]; uint2 u; } o;
        o.h[0] = f2bf(v.x); o.h[1] = f2bf(v.y);
        o.h[2] = f2bf(v.z); o.h[3] = f2bf(v.w);
        *reinterpret_cast<uint2*>(xb + i) = o.u;
    }
}

// src: fp32 [rows][cols]  ->  dst: bf16 [cols][rows]
__global__ void __launch_bounds__(256) transpose_bf16_kernel(const float* __restrict__ src,
                                                             __bf16* __restrict__ dst,
                                                             int rows, int cols) {
    __shared__ float tile[32][33];
    int c0 = blockIdx.x * 32;
    int r0 = blockIdx.y * 32;
    int tx = threadIdx.x & 31;
    int ty = threadIdx.x >> 5;                       // 0..7
    #pragma unroll
    for (int i = 0; i < 32; i += 8)
        tile[ty + i][tx] = src[(size_t)(r0 + ty + i) * cols + c0 + tx];
    __syncthreads();
    #pragma unroll
    for (int i = 0; i < 32; i += 8)
        dst[(size_t)(c0 + ty + i) * rows + r0 + tx] = f2bf(tile[tx][ty + i]);
}

__global__ void __launch_bounds__(256) init_h_kernel(const float* __restrict__ h0,
                                                     float* __restrict__ out0,
                                                     __bf16* __restrict__ hb0, int n) {
    int i = blockIdx.x * 256 + threadIdx.x;
    if (i < n) {
        float v = h0[i];
        out0[i] = v;
        hb0[i]  = f2bf(v);
    }
}

// --------------------------- Phase 1: Wx GEMM ------------------------------
// C[16384][3072] = Xb[16384][1024] @ Kt^T (+bias).
// Block = 4 waves covering 32 rows x 256 cols. Each wave: 32x64 register
// tile = 2 (M) x 4 (N) WMMA sub-tiles; per K-step loads 2 A + 4 B tiles and
// issues 8 v_wmma_f32_16x16x32_bf16.
__global__ void __launch_bounds__(128) wx_gemm_kernel(const __bf16* __restrict__ Xb,
                                                      const __bf16* __restrict__ Kt,
                                                      const float* __restrict__ bias,
                                                      float* __restrict__ Wx) {
    int lane = threadIdx.x & 31;
    int wave = threadIdx.x >> 5;
    int m0 = blockIdx.x * 32;
    int n0 = blockIdx.y * 256 + wave * 64;

    v8f acc[2][4];
    #pragma unroll
    for (int i = 0; i < 2; i++)
        #pragma unroll
        for (int j = 0; j < 4; j++)
            acc[i][j] = (v8f){};

    #pragma unroll 2
    for (int k0 = 0; k0 < D; k0 += 32) {
        v16bf a0 = load_tile_A(Xb, D, m0,      k0, lane);
        v16bf a1 = load_tile_A(Xb, D, m0 + 16, k0, lane);
        #pragma unroll
        for (int j = 0; j < 4; j++) {
            v16bf b = load_tile_B(Kt, D, n0 + j * 16, k0, lane);
            acc[0][j] = __builtin_amdgcn_wmma_f32_16x16x32_bf16(false, a0, false, b,
                                                                (short)0, acc[0][j], false, false);
            acc[1][j] = __builtin_amdgcn_wmma_f32_16x16x32_bf16(false, a1, false, b,
                                                                (short)0, acc[1][j], false, false);
        }
    }

    int mb = ((lane >> 4) << 3);                     // 0 or 8 within sub-tile
    #pragma unroll
    for (int j = 0; j < 4; j++) {
        int   n  = n0 + j * 16 + (lane & 15);
        float bv = bias[n];
        #pragma unroll
        for (int i = 0; i < 2; i++) {
            int mbase = m0 + i * 16 + mb;
            #pragma unroll
            for (int r = 0; r < 8; r++)
                Wx[(size_t)(mbase + r) * G3 + n] = acc[i][j][r] + bv;
        }
    }
}

// --------------------------- Phase 2: GRU step -----------------------------
// One block per (16 batch rows x 16 hidden cols). 4 waves split K=1024; each
// wave computes z/r/g gate tiles (24 WMMAs), partials reduced through LDS.
__global__ void __launch_bounds__(128) gru_step_kernel(const __bf16* __restrict__ hb_prev,
                                                       __bf16* __restrict__ hb_next,
                                                       const float* __restrict__ h_prev,
                                                       float* __restrict__ h_out,
                                                       const float* __restrict__ Wx_t,
                                                       const __bf16* __restrict__ Rt,
                                                       const float* __restrict__ rbias) {
    __shared__ float red[4][3][256];
    int lane = threadIdx.x & 31;
    int wave = threadIdx.x >> 5;
    int m0 = blockIdx.x * 16;                        // batch rows
    int n0 = blockIdx.y * 16;                        // hidden cols
    v8f acc0 = {}, acc1 = {}, acc2 = {};
    int kbeg = wave * (H / 4);
    #pragma unroll 2
    for (int k0 = kbeg; k0 < kbeg + (H / 4); k0 += 32) {
        v16bf a  = load_tile_A(hb_prev, H, m0, k0, lane);
        v16bf b0 = load_tile_B(Rt, H, n0,          k0, lane);
        acc0 = __builtin_amdgcn_wmma_f32_16x16x32_bf16(false, a, false, b0,
                                                       (short)0, acc0, false, false);
        v16bf b1 = load_tile_B(Rt, H, H + n0,      k0, lane);
        acc1 = __builtin_amdgcn_wmma_f32_16x16x32_bf16(false, a, false, b1,
                                                       (short)0, acc1, false, false);
        v16bf b2 = load_tile_B(Rt, H, 2 * H + n0,  k0, lane);
        acc2 = __builtin_amdgcn_wmma_f32_16x16x32_bf16(false, a, false, b2,
                                                       (short)0, acc2, false, false);
    }
    int n  = lane & 15;
    int mb = (lane >> 4) << 3;
    #pragma unroll
    for (int i = 0; i < 8; i++) {
        red[wave][0][(mb + i) * 16 + n] = acc0[i];
        red[wave][1][(mb + i) * 16 + n] = acc1[i];
        red[wave][2][(mb + i) * 16 + n] = acc2[i];
    }
    __syncthreads();
    #pragma unroll
    for (int e = 0; e < 2; e++) {
        int idx = threadIdx.x * 2 + e;               // 0..255 over the 16x16 tile
        int m  = idx >> 4;
        int nn = idx & 15;
        float zv = red[0][0][idx] + red[1][0][idx] + red[2][0][idx] + red[3][0][idx];
        float rv = red[0][1][idx] + red[1][1][idx] + red[2][1][idx] + red[3][1][idx];
        float gv = red[0][2][idx] + red[1][2][idx] + red[2][2][idx] + red[3][2][idx];
        int gb  = m0 + m;
        int col = n0 + nn;
        const float* wrow = Wx_t + (size_t)gb * G3;
        float z = sigmoidf(wrow[col]         + zv + rbias[col]);
        float r = sigmoidf(wrow[H + col]     + rv + rbias[H + col]);
        float g = tanhf(   wrow[2 * H + col] + r * (gv + rbias[2 * H + col]));
        float hp = h_prev[(size_t)gb * H + col];
        float hn = z * hp + (1.0f - z) * g;
        h_out[(size_t)gb * H + col]   = hn;
        hb_next[(size_t)gb * H + col] = f2bf(hn);
    }
}

// ------------------------------ Launcher -----------------------------------

extern "C" void kernel_launch(void* const* d_in, const int* in_sizes, int n_in,
                              void* d_out, int out_size, void* d_ws, size_t ws_size,
                              hipStream_t stream) {
    const float* x     = (const float*)d_in[0];   // [T][B][D]
    const float* h0    = (const float*)d_in[1];   // [B][H]
    const float* kern  = (const float*)d_in[2];   // [D][3H]
    const float* rkern = (const float*)d_in[3];   // [H][3H]
    const float* bias  = (const float*)d_in[4];   // [3H]
    const float* rbias = (const float*)d_in[5];   // [3H]
    float* out = (float*)d_out;                   // [T+1][B][H]

    char* ws = (char*)d_ws;
    size_t off = 0;
    float*  Wx  = (float*)(ws + off);  off += (size_t)T * Bb * G3 * sizeof(float);   // 192 MiB
    __bf16* Xb  = (__bf16*)(ws + off); off += (size_t)T * Bb * D * sizeof(__bf16);   //  32 MiB
    __bf16* Kt  = (__bf16*)(ws + off); off += (size_t)G3 * D * sizeof(__bf16);       //   6 MiB
    __bf16* Rt  = (__bf16*)(ws + off); off += (size_t)G3 * H * sizeof(__bf16);       //   6 MiB
    __bf16* hb0 = (__bf16*)(ws + off); off += (size_t)Bb * H * sizeof(__bf16);
    __bf16* hb1 = (__bf16*)(ws + off); off += (size_t)Bb * H * sizeof(__bf16);
    __bf16* hb[2] = { hb0, hb1 };

    // Phase 0: precision conversion + weight transposes + h0 init.
    {
        int n = T * Bb * D;                                    // 16,777,216
        convert_x_kernel<<<n / (256 * 4), 256, 0, stream>>>(x, Xb, n);
    }
    transpose_bf16_kernel<<<dim3(G3 / 32, D / 32), 256, 0, stream>>>(kern,  Kt, D, G3);
    transpose_bf16_kernel<<<dim3(G3 / 32, H / 32), 256, 0, stream>>>(rkern, Rt, H, G3);
    init_h_kernel<<<(Bb * H) / 256, 256, 0, stream>>>(h0, out, hb0, Bb * H);

    // Phase 1: Wx = x @ kernel + bias  (32-row x 256-col blocks, 4 waves).
    wx_gemm_kernel<<<dim3((T * Bb) / 32, G3 / 256), 128, 0, stream>>>(Xb, Kt, bias, Wx);

    // Phase 2: sequential scan, one kernel per timestep.
    for (int t = 0; t < T; t++) {
        const __bf16* hprev_b = hb[t & 1];
        __bf16*       hnext_b = hb[(t + 1) & 1];
        const float*  hprev_f = out + (size_t)t * Bb * H;
        float*        hnext_f = out + (size_t)(t + 1) * Bb * H;
        const float*  Wxt     = Wx + (size_t)t * Bb * G3;
        gru_step_kernel<<<dim3(Bb / 16, H / 16), 128, 0, stream>>>(
            hprev_b, hnext_b, hprev_f, hnext_f, Wxt, Rt, rbias);
    }
}
